// Pooling2nd_37838661878411
// MI455X (gfx1250) — matrix operand order repaired and checked by reading, not compiled, not run on hardware
//
#include <hip/hip_runtime.h>
#include <hip/hip_bf16.h>
#include <math.h>

// ---------------------------------------------------------------------------
// MI455X / gfx1250 implementation.
// x: (4,64,128,128) f32, mask: (4,1,128,128) i32 -> out: (4,1,128,128) f32
//
// Correlation phase is expressed as a 16x112x64 GEMM per 16-pixel row tile
// using v_wmma_f32_16x16x32_bf16 with bf16(hi)+bf16(lo) split inputs
// (hi*hi + hi*lo + lo*hi) for fp32-class accuracy on the matrix pipes.
// Workspace use: 4 arrays of 4*128*128*64 u16 = 32 MiB total.
// ---------------------------------------------------------------------------

typedef __attribute__((ext_vector_type(16))) __bf16         v16bf;
typedef __attribute__((ext_vector_type(8)))  float          v8f;
typedef __attribute__((ext_vector_type(8)))  unsigned short us8;

#define NB   4
#define NC   64
#define NH   128
#define NW   128
#define NPIX (NB * NH * NW)          // 65536
#define EPS  1e-5f

// fp32 -> bf16 bits, round-to-nearest-even
__device__ __forceinline__ unsigned short f2bf(float f) {
    unsigned u = __builtin_bit_cast(unsigned, f);
    unsigned r = u + 0x7FFFu + ((u >> 16) & 1u);
    return (unsigned short)(r >> 16);
}
__device__ __forceinline__ float bf2f(unsigned short s) {
    unsigned u = ((unsigned)s) << 16;
    return __builtin_bit_cast(float, u);
}
// jnp.pad 'reflect' index map for [0,127]
__device__ __forceinline__ int refl(int t) {
    t = t < 0 ? -t : t;
    return t > 127 ? 254 - t : t;
}

// ---------------------------------------------------------------------------
// Kernel 1: channel-center + L2-normalize; emit bf16 hi/lo, channel-last.
// ---------------------------------------------------------------------------
__global__ __launch_bounds__(256) void k_normalize(
        const float* __restrict__ x,
        unsigned short* __restrict__ xnh,
        unsigned short* __restrict__ xnl) {
    int t  = blockIdx.x * blockDim.x + threadIdx.x;   // pixel id, exactly 65536
    int b  = t >> 14;
    int hw = t & 16383;
    const float* xp = x + (size_t)b * NC * NH * NW + hw;

    float v[NC];
    float s = 0.f;
    #pragma unroll
    for (int c = 0; c < NC; ++c) { v[c] = xp[c * NH * NW]; s += v[c]; }
    float mean = s * (1.0f / NC);
    float ss = 0.f;
    #pragma unroll
    for (int c = 0; c < NC; ++c) { v[c] -= mean; ss += v[c] * v[c]; }
    float inv = 1.0f / (sqrtf(ss) + EPS);

    us8* oh = (us8*)(xnh + (size_t)t * NC);
    us8* ol = (us8*)(xnl + (size_t)t * NC);
    #pragma unroll
    for (int i = 0; i < NC / 8; ++i) {
        us8 hv, lv;
        #pragma unroll
        for (int j = 0; j < 8; ++j) {
            float xv = v[i * 8 + j] * inv;
            unsigned short hb = f2bf(xv);
            hv[j] = hb;
            lv[j] = f2bf(xv - bf2f(hb));
        }
        oh[i] = hv;
        ol[i] = lv;
    }
}

// ---------------------------------------------------------------------------
// Kernel 2: 3x3 mask-selected mean pool of xn -> xbase (bf16 hi/lo).
// ---------------------------------------------------------------------------
__global__ __launch_bounds__(256) void k_selbase(
        const unsigned short* __restrict__ xnh,
        const unsigned short* __restrict__ xnl,
        const int* __restrict__ mask,
        unsigned short* __restrict__ xbh,
        unsigned short* __restrict__ xbl) {
    int t  = blockIdx.x * blockDim.x + threadIdx.x;
    int b  = t >> 14;
    int hw = t & 16383;
    int h  = hw >> 7, w = hw & 127;
    int mc = mask[t];

    float acc[NC];
    #pragma unroll
    for (int c = 0; c < NC; ++c) acc[c] = 0.f;
    int cnt = 0;

    for (int dy = -1; dy <= 1; ++dy) {
        int rr = refl(h + dy);
        for (int dx = -1; dx <= 1; ++dx) {
            int cc = refl(w + dx);
            int p  = (b << 14) + (rr << 7) + cc;
            if (mask[p] == mc) {
                ++cnt;
                const us8* ph = (const us8*)(xnh + (size_t)p * NC);
                const us8* pl = (const us8*)(xnl + (size_t)p * NC);
                #pragma unroll
                for (int i = 0; i < NC / 8; ++i) {
                    us8 hv = ph[i], lv = pl[i];
                    #pragma unroll
                    for (int j = 0; j < 8; ++j)
                        acc[i * 8 + j] += bf2f(hv[j]) + bf2f(lv[j]);
                }
            }
        }
    }
    float inv = 1.0f / (float)cnt;   // cnt >= 1 (center always matches)
    us8* oh = (us8*)(xbh + (size_t)t * NC);
    us8* ol = (us8*)(xbl + (size_t)t * NC);
    #pragma unroll
    for (int i = 0; i < NC / 8; ++i) {
        us8 hv, lv;
        #pragma unroll
        for (int j = 0; j < 8; ++j) {
            float xv = acc[i * 8 + j] * inv;
            unsigned short hb = f2bf(xv);
            hv[j] = hb;
            lv[j] = f2bf(xv - bf2f(hb));
        }
        oh[i] = hv;
        ol[i] = lv;
    }
}

// ---------------------------------------------------------------------------
// Kernel 3: per 16-pixel row tile, G(16x112) = xbase(16x64) * xn_halo(64x112)
// via v_wmma_f32_16x16x32_bf16 (bf16x3 split), then match/exp/count epilogue.
// 8 waves per block, one tile per wave; grid = b*h = 512 blocks.
// ---------------------------------------------------------------------------
__global__ __launch_bounds__(256) void k_corr(
        const unsigned short* __restrict__ xnh,
        const unsigned short* __restrict__ xnl,
        const unsigned short* __restrict__ xbh,
        const unsigned short* __restrict__ xbl,
        const int* __restrict__ mask,
        float* __restrict__ out) {
    __shared__ float Gs[8][16 * 112];           // 56 KiB, per-wave slabs

    const int wave  = threadIdx.x >> 5;
    const int lane  = threadIdx.x & 31;
    const int b     = blockIdx.x >> 7;
    const int h     = blockIdx.x & 127;
    const int w0    = wave << 4;
    const int nl    = lane & 15;
    const int khalf = lane >> 4;

    union Frag { v16bf v; us8 h2[2]; };

    // --- A fragments: xbase, M = pixel w0+nl, ISA 16-bit A 16x32 layout:
    //     lane<16 holds k in [k0,k0+8) U [k0+16,k0+24) with k0 = 8*(lane>=16)
    const int pa = (b << 14) + (h << 7) + (w0 + nl);
    const int k0 = khalf << 3;
    const unsigned short* pah = xbh + (size_t)pa * NC;
    const unsigned short* pal = xbl + (size_t)pa * NC;
    Frag ah0, al0, ah1, al1;                    // K-step 0: ch 0..31, step 1: ch 32..63
    ah0.h2[0] = *(const us8*)(pah + k0);       ah0.h2[1] = *(const us8*)(pah + k0 + 16);
    al0.h2[0] = *(const us8*)(pal + k0);       al0.h2[1] = *(const us8*)(pal + k0 + 16);
    ah1.h2[0] = *(const us8*)(pah + k0 + 32);  ah1.h2[1] = *(const us8*)(pah + k0 + 48);
    al1.h2[0] = *(const us8*)(pal + k0 + 32);  al1.h2[1] = *(const us8*)(pal + k0 + 48);

    for (int chunk = 0; chunk < 7; ++chunk) {
        // --- B fragments: halo pixel n (5 rows x 20 cols, reflect-mapped),
        //     ISA 16-bit B 32x16 layout: lane holds k in [kb,kb+8) U [kb+16,kb+24) with kb = 8*(lane>=16)
        const int n = (chunk << 4) + nl;
        Frag bh0, bl0, bh1, bl1;
        if (n < 100) {
            const int i  = n / 20, j = n % 20;
            const int rr = refl(h + i - 2), cc = refl(w0 + j - 2);
            const int pb = (b << 14) + (rr << 7) + cc;
            const int kb = khalf << 3;
            const unsigned short* pbh = xnh + (size_t)pb * NC;
            const unsigned short* pbl = xnl + (size_t)pb * NC;
            bh0.h2[0] = *(const us8*)(pbh + kb);       bh0.h2[1] = *(const us8*)(pbh + kb + 16);
            bl0.h2[0] = *(const us8*)(pbl + kb);       bl0.h2[1] = *(const us8*)(pbl + kb + 16);
            bh1.h2[0] = *(const us8*)(pbh + kb + 32);  bh1.h2[1] = *(const us8*)(pbh + kb + 48);
            bl1.h2[0] = *(const us8*)(pbl + kb + 32);  bl1.h2[1] = *(const us8*)(pbl + kb + 48);
        } else {
            us8 z = {0, 0, 0, 0, 0, 0, 0, 0};   // pad columns, never read back
            bh0.h2[0] = z; bh0.h2[1] = z; bl0.h2[0] = z; bl0.h2[1] = z;
            bh1.h2[0] = z; bh1.h2[1] = z; bl1.h2[0] = z; bl1.h2[1] = z;
        }

        v8f c = {};
        // K-step 0 (channels 0..31): hi*hi + hi*lo + lo*hi
        c = __builtin_amdgcn_wmma_f32_16x16x32_bf16(false, ah0.v, false, bh0.v, (short)0, c, false, false);
        c = __builtin_amdgcn_wmma_f32_16x16x32_bf16(false, ah0.v, false, bl0.v, (short)0, c, false, false);
        c = __builtin_amdgcn_wmma_f32_16x16x32_bf16(false, al0.v, false, bh0.v, (short)0, c, false, false);
        // K-step 1 (channels 32..63)
        c = __builtin_amdgcn_wmma_f32_16x16x32_bf16(false, ah1.v, false, bh1.v, (short)0, c, false, false);
        c = __builtin_amdgcn_wmma_f32_16x16x32_bf16(false, ah1.v, false, bl1.v, (short)0, c, false, false);
        c = __builtin_amdgcn_wmma_f32_16x16x32_bf16(false, al1.v, false, bh1.v, (short)0, c, false, false);

        // C/D layout: VGPR r -> M = r + 8*(lane>=16), N = lane%16
        #pragma unroll
        for (int r = 0; r < 8; ++r)
            Gs[wave][(r + (khalf << 3)) * 112 + (chunk << 4) + nl] = c[r];
    }

    __syncthreads();

    // --- Epilogue: lanes 0..15, one output pixel each.
    if (lane < 16) {
        const int m    = lane;
        const int w    = w0 + m;
        const int base = (b << 14) + (h << 7);
        const int mc   = mask[base + w];
        float acc = 0.f;
        int   cnt = 0;
        #pragma unroll
        for (int dy = -2; dy <= 2; ++dy) {
            const int rr = refl(h + dy);
            #pragma unroll
            for (int dx = -2; dx <= 2; ++dx) {
                const int cc = refl(w + dx);
                const bool mt = (mask[(b << 14) + (rr << 7) + cc] == mc);
                cnt += mt ? 1 : 0;
                const float g = Gs[wave][m * 112 + (dy + 2) * 20 + (m + dx + 2)];
                acc += mt ? expf(g) : 1.0f;   // exp(match*corr): exp(0)=1 when unmatched
            }
        }
        out[base + w] = acc / (float)cnt;     // cnt >= 1 always
    }
}

// ---------------------------------------------------------------------------
extern "C" void kernel_launch(void* const* d_in, const int* in_sizes, int n_in,
                              void* d_out, int out_size, void* d_ws, size_t ws_size,
                              hipStream_t stream) {
    const float* x    = (const float*)d_in[0];   // (4,64,128,128) f32
    const int*   mask = (const int*)d_in[1];     // (4,1,128,128)  i32
    float*       out  = (float*)d_out;           // (4,1,128,128)  f32

    const size_t NELEM = (size_t)NPIX * NC;      // 4,194,304 u16 per array
    unsigned short* xnh = (unsigned short*)d_ws;         // needs 32 MiB total ws
    unsigned short* xnl = xnh + NELEM;
    unsigned short* xbh = xnl + NELEM;
    unsigned short* xbl = xbh + NELEM;

    k_normalize<<<NPIX / 256, 256, 0, stream>>>(x, xnh, xnl);
    k_selbase  <<<NPIX / 256, 256, 0, stream>>>(xnh, xnl, mask, xbh, xbl);
    k_corr     <<<NB * NH,    256, 0, stream>>>(xnh, xnl, xbh, xbl, mask, out);
}
